// GraphFPModule_11690900979988
// MI455X (gfx1250) — matrix operand order repaired and checked by reading, not compiled, not run on hardware
//
#include <hip/hip_runtime.h>
#include <hip/hip_bf16.h>

// ---------------------------------------------------------------------------
// MI455X (gfx1250) implementation.
//
// Heavy path: per-edge radial-head GEMMs (E=16384, K=32, Ntot=6144) fused with
// the equivariant basis contraction so the 400MB R tensor is never
// materialized. One workgroup (8 wave32) per 16-edge tile; each wave runs
// v_wmma_f32_16x16x32_f16 over (weight-col-tile x edge-tile) jobs and folds
// each accumulator straight into LDS msg buffers via ds_add_f32, then the
// workgroup scatters to node accumulators with global_atomic_add_f32.
// The four radial heads are statically specialized loops so the B fragment is
// a fixed register operand (no per-iteration cndmask selection) and the
// per-head epilogue is branch-free.
// ---------------------------------------------------------------------------

typedef __attribute__((ext_vector_type(16))) _Float16 v16h;
typedef __attribute__((ext_vector_type(8)))  float    v8f;

#define BSZ    2
#define NPTS   512
#define MPREV  128
#define EDGES  16384
#define NNODE  1024          // BSZ*NPTS
#define NJOBS  384           // 64 (head00) + 64 (head01) + 64 (head10) + 192 (head11)

// ---------------- workspace layout (floats) --------------------------------
#define WS_F0      0
#define WS_F1      (WS_F0 + NNODE*32)
#define WS_HH      (WS_F1 + NNODE*32*3)
#define WS_OUT0    (WS_HH + 4*EDGES*32)
#define WS_OUT1    (WS_OUT0 + NNODE*32)
#define WS_DEG     (WS_OUT1 + NNODE*32*3)
#define WS_W3PACK  (WS_DEG + NNODE)
#define ZERO_LEN   (NNODE*32 + NNODE*32*3 + NNODE)   // out0acc..deg contiguous

// ---------------------------------------------------------------------------
__global__ void k_zero(float* p, int n) {
  int i = blockIdx.x * blockDim.x + threadIdx.x;
  if (i < n) p[i] = 0.0f;
}

// ---------------------------------------------------------------------------
// three_nn + inverse-square-distance interpolation; writes fused features
// f0[n][32] (interp c<16, skip c>=16) and f1[n][32][3].
__global__ void k_interp(const float* __restrict__ h0, const float* __restrict__ h1,
                         const float* __restrict__ uph0, const float* __restrict__ uph1,
                         const float* __restrict__ xyz, const float* __restrict__ xyz_prev,
                         float* __restrict__ f0, float* __restrict__ f1) {
  int n = blockIdx.x * blockDim.x + threadIdx.x;
  if (n >= NNODE) return;
  int b = n / NPTS;
  float px = xyz[n * 3 + 0], py = xyz[n * 3 + 1], pz = xyz[n * 3 + 2];
  float bd0 = 1e30f, bd1 = 1e30f, bd2 = 1e30f;
  int bi0 = b * MPREV, bi1 = b * MPREV, bi2 = b * MPREV;
  for (int j = 0; j < MPREV; ++j) {
    int pidx = b * MPREV + j;
    float dx = px - xyz_prev[pidx * 3 + 0];
    float dy = py - xyz_prev[pidx * 3 + 1];
    float dz = pz - xyz_prev[pidx * 3 + 2];
    float d2 = dx * dx + dy * dy + dz * dz;
    if (d2 < bd0) { bd2 = bd1; bi2 = bi1; bd1 = bd0; bi1 = bi0; bd0 = d2; bi0 = pidx; }
    else if (d2 < bd1) { bd2 = bd1; bi2 = bi1; bd1 = d2; bi1 = pidx; }
    else if (d2 < bd2) { bd2 = d2; bi2 = pidx; }
  }
  float i0 = 1.0f / (fmaxf(bd0, 1e-10f) + 1e-8f);
  float i1 = 1.0f / (fmaxf(bd1, 1e-10f) + 1e-8f);
  float i2 = 1.0f / (fmaxf(bd2, 1e-10f) + 1e-8f);
  float s = i0 + i1 + i2;
  float w0 = i0 / s, w1 = i1 / s, w2 = i2 / s;

  for (int c = 0; c < 16; ++c) {
    f0[n * 32 + c] = w0 * uph0[bi0 * 16 + c] + w1 * uph0[bi1 * 16 + c] + w2 * uph0[bi2 * 16 + c];
    f0[n * 32 + 16 + c] = h0[n * 16 + c];
    for (int m = 0; m < 3; ++m) {
      f1[(n * 32 + c) * 3 + m] = w0 * uph1[(bi0 * 16 + c) * 3 + m]
                               + w1 * uph1[(bi1 * 16 + c) * 3 + m]
                               + w2 * uph1[(bi2 * 16 + c) * 3 + m];
      f1[(n * 32 + 16 + c) * 3 + m] = h1[(n * 16 + c) * 3 + m];
    }
  }
}

// ---------------------------------------------------------------------------
// Pre-pack the four radial-head weight matrices into the f16 WMMA A-fragment
// lane layout: lane L holds row M=L&15, K = half*16 + i (i = 0..15).
__global__ void k_pack_w3(const float* __restrict__ w300, const float* __restrict__ w301,
                          const float* __restrict__ w310, const float* __restrict__ w311,
                          _Float16* __restrict__ pack) {
  int g = blockIdx.x * blockDim.x + threadIdx.x;   // job*32 + lane
  if (g >= NJOBS * 32) return;
  int job = g >> 5, lane = g & 31;
  int p, t;
  if (job < 64)       { p = 0; t = job; }
  else if (job < 128) { p = 1; t = job - 64; }
  else if (job < 192) { p = 2; t = job - 128; }
  else                { p = 3; t = job - 192; }
  const float* w3 = (p == 0) ? w300 : (p == 1) ? w301 : (p == 2) ? w310 : w311;
  int m = lane & 15, half = lane >> 4;
  const float* row = w3 + (t * 16 + m) * 32;
  _Float16* dst = pack + (size_t)job * 512 + lane * 16;
#pragma unroll
  for (int i = 0; i < 8; ++i) dst[i] = (_Float16)row[half * 16 + i];
#pragma unroll
  for (int i = 0; i < 8; ++i) dst[8 + i] = (_Float16)row[half * 16 + 8 + i];
}

// ---------------------------------------------------------------------------
// Radial MLP trunk: hh[p][e][32] = relu(ln2(W2 relu(ln1(W1 r + b1)) + b2)).
__global__ void k_radial(const float* __restrict__ r,
                         const float* __restrict__ w1, const float* __restrict__ b1,
                         const float* __restrict__ g1, const float* __restrict__ n1,
                         const float* __restrict__ w2, const float* __restrict__ b2,
                         const float* __restrict__ g2, const float* __restrict__ n2,
                         float* __restrict__ hh) {
  __shared__ float w2s[32 * 32];
  __shared__ float w1s[32], b1s[32], g1s[32], n1s[32], b2s[32], g2s[32], n2s[32];
  int p = blockIdx.y;
  int tid = threadIdx.x;
  for (int i = tid; i < 1024; i += 256) w2s[i] = w2[p * 1024 + i];
  if (tid < 32) {
    w1s[tid] = w1[p * 32 + tid]; b1s[tid] = b1[p * 32 + tid];
    g1s[tid] = g1[p * 32 + tid]; n1s[tid] = n1[p * 32 + tid];
    b2s[tid] = b2[p * 32 + tid]; g2s[tid] = g2[p * 32 + tid];
    n2s[tid] = n2[p * 32 + tid];
  }
  __syncthreads();
  int e = blockIdx.x * 256 + tid;
  if (e >= EDGES) return;
  float rv = r[e];
  float h[32];
#pragma unroll
  for (int k = 0; k < 32; ++k) h[k] = w1s[k] * rv + b1s[k];
  float mu = 0.f;
#pragma unroll
  for (int k = 0; k < 32; ++k) mu += h[k];
  mu *= (1.0f / 32.0f);
  float var = 0.f;
#pragma unroll
  for (int k = 0; k < 32; ++k) { float d = h[k] - mu; var += d * d; }
  var *= (1.0f / 32.0f);
  float inv = rsqrtf(var + 1e-5f);
#pragma unroll
  for (int k = 0; k < 32; ++k) h[k] = fmaxf((h[k] - mu) * inv * g1s[k] + n1s[k], 0.f);

  float h2[32];
#pragma unroll
  for (int o = 0; o < 32; ++o) {
    float a = b2s[o];
#pragma unroll
    for (int k = 0; k < 32; ++k) a += w2s[o * 32 + k] * h[k];
    h2[o] = a;
  }
  mu = 0.f;
#pragma unroll
  for (int k = 0; k < 32; ++k) mu += h2[k];
  mu *= (1.0f / 32.0f);
  var = 0.f;
#pragma unroll
  for (int k = 0; k < 32; ++k) { float d = h2[k] - mu; var += d * d; }
  var *= (1.0f / 32.0f);
  inv = rsqrtf(var + 1e-5f);
  float* dst = hh + ((size_t)p * EDGES + e) * 32;
#pragma unroll
  for (int k = 0; k < 32; ++k)
    dst[k] = fmaxf((h2[k] - mu) * inv * g2s[k] + n2s[k], 0.f);
}

// ---------------------------------------------------------------------------
__global__ void k_deg(const int* __restrict__ edge_dst, float* __restrict__ deg) {
  int e = blockIdx.x * blockDim.x + threadIdx.x;
  if (e < EDGES) atomicAdd(&deg[edge_dst[e]], 1.0f);
}

// ---------------------------------------------------------------------------
// Fused WMMA message kernel: one workgroup (8 wave32) per 16-edge tile.
__global__ void __launch_bounds__(256)
k_msg(const float* __restrict__ hh, const _Float16* __restrict__ w3pack,
      const float* __restrict__ b300, const float* __restrict__ b301,
      const float* __restrict__ b310, const float* __restrict__ b311,
      const float* __restrict__ f0, const float* __restrict__ f1,
      const float* __restrict__ basis00, const float* __restrict__ basis01,
      const float* __restrict__ basis10, const float* __restrict__ basis11,
      const float* __restrict__ self_w0, const float* __restrict__ self_w1,
      const int* __restrict__ edge_src, const int* __restrict__ edge_dst,
      float* __restrict__ out0acc, float* __restrict__ out1acc) {
  __shared__ float s0s[16 * 32];        // src0 features
  __shared__ float u1s[16 * 32];        // sum_n basis10[n] * src1[c][n]
  __shared__ float T1s[16 * 32 * 9];    // T1[e][c][m][f]
  __shared__ float d0s[16 * 32];        // dst0
  __shared__ float d1s[16 * 32 * 3];    // dst1
  __shared__ float msg0s[16 * 32];
  __shared__ float msg1s[16 * 32 * 3];
  __shared__ float b00s[16];
  __shared__ float b01s[16 * 3];
  __shared__ int srcs[16], dsts[16];

  int tid = threadIdx.x;
  int ebase = blockIdx.x * 16;

  if (tid < 16) {
    srcs[tid] = edge_src[ebase + tid];
    dsts[tid] = edge_dst[ebase + tid];
    b00s[tid] = basis00[ebase + tid];
  }
  if (tid < 48) b01s[tid] = basis01[ebase * 3 + tid];
  __syncthreads();

  for (int idx = tid; idx < 512; idx += 256) {
    int i = idx >> 5, c = idx & 31;
    s0s[idx] = f0[srcs[i] * 32 + c];
    d0s[idx] = f0[dsts[i] * 32 + c];
    const float* b10 = basis10 + (ebase + i) * 3;
    const float* sf = f1 + srcs[i] * 96 + c * 3;
    u1s[idx] = b10[0] * sf[0] + b10[1] * sf[1] + b10[2] * sf[2];
  }
  for (int idx = tid; idx < 1536; idx += 256) {
    int i = idx / 96;
    d1s[idx] = f1[dsts[i] * 96 + (idx % 96)];
  }
  for (int idx = tid; idx < 4608; idx += 256) {
    int i = idx / 288, rem = idx % 288;
    int c = rem / 9, mf = rem % 9, m = mf / 3, f = mf % 3;
    const float* b11 = basis11 + (ebase + i) * 27;
    const float* sf = f1 + srcs[i] * 96 + c * 3;
    T1s[idx] = b11[(m * 3 + 0) * 3 + f] * sf[0]
             + b11[(m * 3 + 1) * 3 + f] * sf[1]
             + b11[(m * 3 + 2) * 3 + f] * sf[2];
  }
  __syncthreads();

  // msg init = per-edge self interaction on destination features
  for (int idx = tid; idx < 512; idx += 256) {
    int i = idx >> 5, o = idx & 31;
    float a = 0.f;
#pragma unroll
    for (int c = 0; c < 32; ++c) a += self_w0[o * 32 + c] * d0s[i * 32 + c];
    msg0s[idx] = a;
  }
  for (int idx = tid; idx < 1536; idx += 256) {
    int i = idx / 96, rem = idx % 96, o = rem / 3, m = rem % 3;
    float a = 0.f;
#pragma unroll
    for (int c = 0; c < 32; ++c) a += self_w1[o * 32 + c] * d1s[(i * 32 + c) * 3 + m];
    msg1s[idx] = a;
  }

  // Build the four B fragments (hh^T, K x 16 edges): lane L holds edge
  // n=L&15, K = (L>>4)*16 + i  (contiguous 16 floats of hh -> f16).
  int lane = tid & 31, wave = tid >> 5;
  int n = lane & 15, half = lane >> 4;
  v16h Bf0, Bf1, Bf2, Bf3;
  {
    const float* s0 = hh + ((size_t)0 * EDGES + (ebase + n)) * 32 + half * 16;
    const float* s1 = hh + ((size_t)1 * EDGES + (ebase + n)) * 32 + half * 16;
    const float* s2 = hh + ((size_t)2 * EDGES + (ebase + n)) * 32 + half * 16;
    const float* s3 = hh + ((size_t)3 * EDGES + (ebase + n)) * 32 + half * 16;
#pragma unroll
    for (int i = 0; i < 16; ++i) {
      Bf0[i] = (_Float16)s0[i];
      Bf1[i] = (_Float16)s1[i];
      Bf2[i] = (_Float16)s2[i];
      Bf3[i] = (_Float16)s3[i];
    }
  }
  __syncthreads();

  // Statically specialized per-head job loops (p is compile-time; B operand
  // is a fixed register set; EXEC stays all-ones; prefetch is branch-free).
  const int laneoff = lane * 16;

  // ---- head 00 : R00 (E,32o,32c), msg0 += basis00 * sum_c R*src0 ----
  for (int t = wave; t < 64; t += 8) {
    v16h Af = *(const v16h*)(w3pack + (size_t)(0 + t) * 512 + laneoff);
    int tn = (t + 8 < 64) ? (t + 8) : t;
    __builtin_prefetch(w3pack + (size_t)(0 + tn) * 512 + laneoff, 0, 1);
    v8f acc = {};
    acc = __builtin_amdgcn_wmma_f32_16x16x32_f16(false, Af, false, Bf0,
                                                 (short)0, acc, false, false);
    int colbase = t * 16 + half * 8;
    int cbase = ((t & 1) << 4) + half * 8;
    int o = t >> 1;
    float s = 0.f;
#pragma unroll
    for (int j = 0; j < 8; ++j)
      s += (acc[j] + b300[colbase + j]) * s0s[n * 32 + cbase + j];
    atomicAdd(&msg0s[n * 32 + o], b00s[n] * s);
  }

  // ---- head 01 : R01 (E,32o,32c), msg1[m] += basis01[m] * sum_c R*src0 ----
  for (int t = wave; t < 64; t += 8) {
    v16h Af = *(const v16h*)(w3pack + (size_t)(64 + t) * 512 + laneoff);
    int tn = (t + 8 < 64) ? (t + 8) : t;
    __builtin_prefetch(w3pack + (size_t)(64 + tn) * 512 + laneoff, 0, 1);
    v8f acc = {};
    acc = __builtin_amdgcn_wmma_f32_16x16x32_f16(false, Af, false, Bf1,
                                                 (short)0, acc, false, false);
    int colbase = t * 16 + half * 8;
    int cbase = ((t & 1) << 4) + half * 8;
    int o = t >> 1;
    float s = 0.f;
#pragma unroll
    for (int j = 0; j < 8; ++j)
      s += (acc[j] + b301[colbase + j]) * s0s[n * 32 + cbase + j];
    float* md = &msg1s[(n * 32 + o) * 3];
    atomicAdd(&md[0], b01s[n * 3 + 0] * s);
    atomicAdd(&md[1], b01s[n * 3 + 1] * s);
    atomicAdd(&md[2], b01s[n * 3 + 2] * s);
  }

  // ---- head 10 : R10 (E,32o,32c), msg0 += sum_c R * u1 ----
  for (int t = wave; t < 64; t += 8) {
    v16h Af = *(const v16h*)(w3pack + (size_t)(128 + t) * 512 + laneoff);
    int tn = (t + 8 < 64) ? (t + 8) : t;
    __builtin_prefetch(w3pack + (size_t)(128 + tn) * 512 + laneoff, 0, 1);
    v8f acc = {};
    acc = __builtin_amdgcn_wmma_f32_16x16x32_f16(false, Af, false, Bf2,
                                                 (short)0, acc, false, false);
    int colbase = t * 16 + half * 8;
    int cbase = ((t & 1) << 4) + half * 8;
    int o = t >> 1;
    float s = 0.f;
#pragma unroll
    for (int j = 0; j < 8; ++j)
      s += (acc[j] + b310[colbase + j]) * u1s[n * 32 + cbase + j];
    atomicAdd(&msg0s[n * 32 + o], s);
  }

  // ---- head 11 : R11 (E,32o,32c,3f), msg1[m] += sum_{c,f} R * T1[c][m][f] ----
  for (int t = wave; t < 192; t += 8) {
    v16h Af = *(const v16h*)(w3pack + (size_t)(192 + t) * 512 + laneoff);
    int tn = (t + 8 < 192) ? (t + 8) : t;
    __builtin_prefetch(w3pack + (size_t)(192 + tn) * 512 + laneoff, 0, 1);
    v8f acc = {};
    acc = __builtin_amdgcn_wmma_f32_16x16x32_f16(false, Af, false, Bf3,
                                                 (short)0, acc, false, false);
    int colbase = t * 16 + half * 8;
    int o = t / 6;                       // 96 cols per o, 6 tiles per o
    int rembase = colbase - o * 96;      // (c*3+f) of first owned column
    float v0 = 0.f, v1 = 0.f, v2 = 0.f;
#pragma unroll
    for (int j = 0; j < 8; ++j) {
      float R = acc[j] + b311[colbase + j];
      int rem = rembase + j;
      int c = rem / 3, f = rem - c * 3;
      const float* T = &T1s[(n * 32 + c) * 9 + f];
      v0 += R * T[0];
      v1 += R * T[3];
      v2 += R * T[6];
    }
    float* md = &msg1s[(n * 32 + o) * 3];
    atomicAdd(&md[0], v0);
    atomicAdd(&md[1], v1);
    atomicAdd(&md[2], v2);
  }
  __syncthreads();

  // Scatter messages to node accumulators.
  for (int idx = tid; idx < 512; idx += 256) {
    int i = idx >> 5, o = idx & 31;
    atomicAdd(&out0acc[dsts[i] * 32 + o], msg0s[idx]);
  }
  for (int idx = tid; idx < 1536; idx += 256) {
    int i = idx / 96;
    atomicAdd(&out1acc[dsts[i] * 96 + (idx % 96)], msg1s[idx]);
  }
}

// ---------------------------------------------------------------------------
// GNormSE3: one wave32 per node, thread = channel.
__global__ void k_gnorm(const float* __restrict__ out0acc, const float* __restrict__ out1acc,
                        const float* __restrict__ deg,
                        const float* __restrict__ g0, const float* __restrict__ lb0,
                        const float* __restrict__ w0, const float* __restrict__ bb0,
                        const float* __restrict__ g1, const float* __restrict__ lb1,
                        const float* __restrict__ w1, const float* __restrict__ bb1,
                        float* __restrict__ out) {
  __shared__ float nr[32];
  __shared__ float rl[32];
  int node = blockIdx.x;
  int o = threadIdx.x;
  float dc = fmaxf(deg[node], 1.0f);

  // degree 0
  float v = out0acc[node * 32 + o] / dc;
  float nrm = fmaxf(fabsf(v), 1e-12f);
  float phase = v / nrm;
  nr[o] = nrm;
  __syncthreads();
  float mu = 0.f;
  for (int c = 0; c < 32; ++c) mu += nr[c];
  mu *= (1.0f / 32.0f);
  float var = 0.f;
  for (int c = 0; c < 32; ++c) { float d = nr[c] - mu; var += d * d; }
  var *= (1.0f / 32.0f);
  float t = (nrm - mu) * rsqrtf(var + 1e-5f) * g0[o] + lb0[o];
  rl[o] = fmaxf(t, 0.f);
  __syncthreads();
  float a = bb0[o];
  for (int c = 0; c < 32; ++c) a += w0[o * 32 + c] * rl[c];
  out[(node * 32 + o) * 4 + 0] = a * phase;
  __syncthreads();

  // degree 1
  float v0 = out1acc[(node * 32 + o) * 3 + 0] / dc;
  float v1 = out1acc[(node * 32 + o) * 3 + 1] / dc;
  float v2 = out1acc[(node * 32 + o) * 3 + 2] / dc;
  nrm = fmaxf(sqrtf(v0 * v0 + v1 * v1 + v2 * v2), 1e-12f);
  nr[o] = nrm;
  __syncthreads();
  mu = 0.f;
  for (int c = 0; c < 32; ++c) mu += nr[c];
  mu *= (1.0f / 32.0f);
  var = 0.f;
  for (int c = 0; c < 32; ++c) { float d = nr[c] - mu; var += d * d; }
  var *= (1.0f / 32.0f);
  t = (nrm - mu) * rsqrtf(var + 1e-5f) * g1[o] + lb1[o];
  rl[o] = fmaxf(t, 0.f);
  __syncthreads();
  a = bb1[o];
  for (int c = 0; c < 32; ++c) a += w1[o * 32 + c] * rl[c];
  float invn = 1.0f / nrm;
  out[(node * 32 + o) * 4 + 1] = a * v0 * invn;
  out[(node * 32 + o) * 4 + 2] = a * v1 * invn;
  out[(node * 32 + o) * 4 + 3] = a * v2 * invn;
}

// ---------------------------------------------------------------------------
extern "C" void kernel_launch(void* const* d_in, const int* in_sizes, int n_in,
                              void* d_out, int out_size, void* d_ws, size_t ws_size,
                              hipStream_t stream) {
  const float* h0        = (const float*)d_in[0];
  const float* h1        = (const float*)d_in[1];
  const float* uph0      = (const float*)d_in[2];
  const float* uph1      = (const float*)d_in[3];
  const float* xyz       = (const float*)d_in[4];
  const float* xyz_prev  = (const float*)d_in[5];
  const float* r         = (const float*)d_in[6];
  const float* basis00   = (const float*)d_in[7];
  const float* basis01   = (const float*)d_in[8];
  const float* basis10   = (const float*)d_in[9];
  const float* basis11   = (const float*)d_in[10];
  const float* rad_w1    = (const float*)d_in[11];
  const float* rad_b1    = (const float*)d_in[12];
  const float* rad_g1    = (const float*)d_in[13];
  const float* rad_n1    = (const float*)d_in[14];
  const float* rad_w2    = (const float*)d_in[15];
  const float* rad_b2    = (const float*)d_in[16];
  const float* rad_g2    = (const float*)d_in[17];
  const float* rad_n2    = (const float*)d_in[18];
  const float* w3_00     = (const float*)d_in[19];
  const float* b3_00     = (const float*)d_in[20];
  const float* w3_01     = (const float*)d_in[21];
  const float* b3_01     = (const float*)d_in[22];
  const float* w3_10     = (const float*)d_in[23];
  const float* b3_10     = (const float*)d_in[24];
  const float* w3_11     = (const float*)d_in[25];
  const float* b3_11     = (const float*)d_in[26];
  const float* self_w0   = (const float*)d_in[27];
  const float* self_w1   = (const float*)d_in[28];
  const float* n_g0      = (const float*)d_in[29];
  const float* n_b0      = (const float*)d_in[30];
  const float* n_w0      = (const float*)d_in[31];
  const float* n_bb0     = (const float*)d_in[32];
  const float* n_g1      = (const float*)d_in[33];
  const float* n_b1      = (const float*)d_in[34];
  const float* n_w1      = (const float*)d_in[35];
  const float* n_bb1     = (const float*)d_in[36];
  const int*   edge_src  = (const int*)d_in[37];
  const int*   edge_dst  = (const int*)d_in[38];
  float* out = (float*)d_out;

  float* ws      = (float*)d_ws;
  float* f0      = ws + WS_F0;
  float* f1      = ws + WS_F1;
  float* hh      = ws + WS_HH;
  float* out0acc = ws + WS_OUT0;
  float* out1acc = ws + WS_OUT1;
  float* deg     = ws + WS_DEG;
  _Float16* w3pack = (_Float16*)(ws + WS_W3PACK);

  k_zero<<<(ZERO_LEN + 255) / 256, 256, 0, stream>>>(out0acc, ZERO_LEN);
  k_pack_w3<<<(NJOBS * 32 + 255) / 256, 256, 0, stream>>>(w3_00, w3_01, w3_10, w3_11, w3pack);
  k_interp<<<(NNODE + 255) / 256, 256, 0, stream>>>(h0, h1, uph0, uph1, xyz, xyz_prev, f0, f1);
  k_radial<<<dim3(EDGES / 256, 4), 256, 0, stream>>>(r, rad_w1, rad_b1, rad_g1, rad_n1,
                                                     rad_w2, rad_b2, rad_g2, rad_n2, hh);
  k_deg<<<EDGES / 256, 256, 0, stream>>>(edge_dst, deg);
  k_msg<<<EDGES / 16, 256, 0, stream>>>(hh, w3pack, b3_00, b3_01, b3_10, b3_11,
                                        f0, f1, basis00, basis01, basis10, basis11,
                                        self_w0, self_w1, edge_src, edge_dst,
                                        out0acc, out1acc);
  k_gnorm<<<NNODE, 32, 0, stream>>>(out0acc, out1acc, deg,
                                    n_g0, n_b0, n_w0, n_bb0,
                                    n_g1, n_b1, n_w1, n_bb1, out);
}
